// DecoderBackbone_17068200035047
// MI455X (gfx1250) — matrix operand order repaired and checked by reading, not compiled, not run on hardware
//
#include <hip/hip_runtime.h>
#include <hip/hip_bf16.h>
#include <math.h>

// ---------------------------------------------------------------------------
// DecoderBackbone (conv stem + 4x Mamba) for MI455X / gfx1250 (wave32, WMMA)
// GEMMs (in_proj / x_proj / out_proj) run on v_wmma_f32_16x16x32_f16 with f32
// accumulate. B operand is staged into LDS by the Tensor Data Mover
// (tensor_load_to_lds + s_wait_tensorcnt); A streams from global with inline
// f32->f16 conversion. Selective scan is a register-resident recurrence.
// ---------------------------------------------------------------------------

typedef __attribute__((ext_vector_type(16))) _Float16 v16h;
typedef __attribute__((ext_vector_type(8)))  float    v8f;
typedef __attribute__((ext_vector_type(4)))  unsigned int v4ui;
typedef __attribute__((ext_vector_type(8)))  int      v8si;
typedef __attribute__((ext_vector_type(4)))  int      v4si;

#define NL    4
#define HID   256
#define DIN   512
#define DST   16
#define DCONV 4
#define DTR   16
#define BATCH 4
#define LSEQ  4096           // 64*64 after two stride-2 convs
#define MROWS (BATCH * LSEQ) // 16384
#define KCH   8              // K-fragment chunk staged per TDM transfer

#if defined(__AMDGCN__) && __has_builtin(__builtin_amdgcn_tensor_load_to_lds) && \
    __has_builtin(__builtin_amdgcn_s_wait_tensorcnt)
#define HAVE_TDM 1
#else
#define HAVE_TDM 0
#endif

__device__ __forceinline__ float sigmoidf_(float x) { return 1.0f / (1.0f + __expf(-x)); }
__device__ __forceinline__ float siluf_(float x)    { return x * sigmoidf_(x); }

// ===========================================================================
// Conv stem
// ===========================================================================

// conv1: [4,3,256,256] -> [4,128,128,128], stride 2, pad 1
__global__ void conv1_k(const float* __restrict__ x, const float* __restrict__ w,
                        const float* __restrict__ bias, float* __restrict__ y) {
    int idx = blockIdx.x * blockDim.x + threadIdx.x;           // 4*128*128*128
    int ow = idx & 127, oh = (idx >> 7) & 127, oc = (idx >> 14) & 127, b = idx >> 21;
    float acc = bias[oc];
    #pragma unroll
    for (int ic = 0; ic < 3; ++ic)
        #pragma unroll
        for (int kh = 0; kh < 3; ++kh) {
            int ih = oh * 2 + kh - 1;
            if (ih < 0 || ih > 255) continue;
            #pragma unroll
            for (int kw = 0; kw < 3; ++kw) {
                int iw = ow * 2 + kw - 1;
                if (iw < 0 || iw > 255) continue;
                acc += x[(((size_t)b * 3 + ic) * 256 + ih) * 256 + iw] *
                       w[((oc * 3 + ic) * 3 + kh) * 3 + kw];
            }
        }
    y[idx] = acc;
}

// conv2: [4,128,128,128] -> [4,256,64,64], stride 2, pad 1
__global__ void conv2_k(const float* __restrict__ x, const float* __restrict__ w,
                        const float* __restrict__ bias, float* __restrict__ y) {
    int idx = blockIdx.x * blockDim.x + threadIdx.x;           // 4*256*64*64
    int ow = idx & 63, oh = (idx >> 6) & 63, oc = (idx >> 12) & 255, b = idx >> 20;
    float acc = bias[oc];
    for (int ic = 0; ic < 128; ++ic) {
        const float* xp = x + (((size_t)b * 128 + ic) * 128) * 128;
        const float* wp = w + ((size_t)oc * 128 + ic) * 9;
        #pragma unroll
        for (int kh = 0; kh < 3; ++kh) {
            int ih = oh * 2 + kh - 1;
            if (ih < 0 || ih > 127) continue;
            #pragma unroll
            for (int kw = 0; kw < 3; ++kw) {
                int iw = ow * 2 + kw - 1;
                if (iw < 0 || iw > 127) continue;
                acc += xp[(size_t)ih * 128 + iw] * wp[kh * 3 + kw];
            }
        }
    }
    y[idx] = acc;
}

// GroupNorm statistics: one block per (batch, group); groups = 8
__global__ void gn_stats_k(const float* __restrict__ y, float* __restrict__ stats,
                           int C, int HW) {
    int bg = blockIdx.x, b = bg >> 3, g = bg & 7;
    int cpg = C >> 3;
    int n = cpg * HW;
    const float* base = y + ((size_t)b * C + (size_t)g * cpg) * HW;
    float s = 0.f, q = 0.f;
    for (int i = threadIdx.x; i < n; i += blockDim.x) { float v = base[i]; s += v; q += v * v; }
    __shared__ float shs[256], shq[256];
    shs[threadIdx.x] = s; shq[threadIdx.x] = q;
    __syncthreads();
    for (int st = 128; st > 0; st >>= 1) {
        if (threadIdx.x < st) { shs[threadIdx.x] += shs[threadIdx.x + st];
                                shq[threadIdx.x] += shq[threadIdx.x + st]; }
        __syncthreads();
    }
    if (threadIdx.x == 0) {
        float m = shs[0] / n;
        float v = shq[0] / n - m * m;
        stats[bg * 2]     = m;
        stats[bg * 2 + 1] = rsqrtf(v + 1e-5f);
    }
}

__global__ void gn_apply_silu_k(float* __restrict__ y, const float* __restrict__ stats,
                                const float* __restrict__ w, const float* __restrict__ b,
                                int C, int HW) {
    int idx = blockIdx.x * blockDim.x + threadIdx.x;   // B*C*HW
    int c  = (idx / HW) % C;
    int bb = idx / (HW * C);
    int g  = c / (C >> 3);
    float m = stats[(bb * 8 + g) * 2], r = stats[(bb * 8 + g) * 2 + 1];
    float v = (y[idx] - m) * r * w[c] + b[c];
    y[idx] = siluf_(v);
}

// [4,256,4096] (b,c,l) -> s[b,l,c]
__global__ void to_seq_k(const float* __restrict__ act, float* __restrict__ s) {
    int idx = blockIdx.x * blockDim.x + threadIdx.x;   // 4*256*4096
    int l = idx & 4095, c = (idx >> 12) & 255, b = idx >> 20;
    s[((size_t)((b << 12) | l)) * HID + c] = act[idx];
}

__global__ void write_hw_k(float* __restrict__ out, int off) {
    if (threadIdx.x == 0) { out[off] = 64.0f; out[off + 1] = 64.0f; }
}

// ===========================================================================
// LayerNorm: one wave per 256-wide row (wave32)
// ===========================================================================
__global__ __launch_bounds__(256) void ln_k(const float* __restrict__ s,
                                            const float* __restrict__ w,
                                            const float* __restrict__ b,
                                            float* __restrict__ xn) {
    int wid = threadIdx.x >> 5, lane = threadIdx.x & 31;
    int row = blockIdx.x * 8 + wid;
    const float* p = s + (size_t)row * HID;
    float v[8], sum = 0.f;
    #pragma unroll
    for (int j = 0; j < 8; ++j) { v[j] = p[j * 32 + lane]; sum += v[j]; }
    #pragma unroll
    for (int m = 16; m > 0; m >>= 1) sum += __shfl_xor(sum, m, 32);
    float mean = sum * (1.0f / HID);
    float q = 0.f;
    #pragma unroll
    for (int j = 0; j < 8; ++j) { float d = v[j] - mean; q += d * d; }
    #pragma unroll
    for (int m = 16; m > 0; m >>= 1) q += __shfl_xor(q, m, 32);
    float rstd = rsqrtf(q * (1.0f / HID) + 1e-5f);
    float* o = xn + (size_t)row * HID;
    #pragma unroll
    for (int j = 0; j < 8; ++j) {
        int c = j * 32 + lane;
        o[c] = (v[j] - mean) * rstd * w[c] + b[c];
    }
}

// ===========================================================================
// WMMA GEMM:  C[M,N] (+)= A[M,K] * W[N,K]^T   (f32 in, f16 matrix ops, f32 acc)
// ===========================================================================

// Pack W[N,K] (f32) into B-fragment order (lane%16 = N, element = K in A's table order),
// zero-padding N up to Npad (multiple of 64). Fragment t = nt*(K/32)+kt holds
// 32 lanes x 16 f16, contiguous (1 KB).
__global__ void pack_w_k(const float* __restrict__ W, _Float16* __restrict__ Bp,
                         int N, int K, int Npad) {
    int idx = blockIdx.x * blockDim.x + threadIdx.x;   // Npad*K
    if (idx >= Npad * K) return;
    int e    = idx & 15;
    int lane = (idx >> 4) & 31;
    int t    = idx >> 9;
    int nkt  = K >> 5;
    int kt   = t % nkt, nt = t / nkt;
    int n = nt * 16 + (lane & 15), k = kt * 32 + (lane >> 4) * 8 + (e & 7) + (e >> 3) * 16;
    float v = (n < N) ? W[(size_t)n * K + k] : 0.0f;
    Bp[idx] = (_Float16)v;
}

// A fragment (16x32 f16) per ISA layout: lane holds K = hi*8+{0..7} and
// K = 16+hi*8+{0..7} where hi = lane>>4 (M repeats across the two halves).
__device__ __forceinline__ v16h load_a_frag(const float* __restrict__ A,
                                            int rowBase, int K, int kBase, int lane) {
    int m  = rowBase + (lane & 15);
    int hi = lane >> 4;
    const float* p = A + (size_t)m * K + kBase + hi * 8;
    v16h a;
    #pragma unroll
    for (int j = 0; j < 8; ++j) a[j]     = (_Float16)p[j];
    #pragma unroll
    for (int j = 0; j < 8; ++j) a[8 + j] = (_Float16)p[16 + j];
    return a;
}

// One wave (32 threads) per 32x64 C tile: 2 M-frags x 4 N-frags, K steps of 32.
// Requires (K/32) % KCH == 0 when the TDM path is enabled (K = 256 or 512 here).
__global__ __launch_bounds__(32) void gemm_wmma_k(const float* __restrict__ A,
                                                  const _Float16* __restrict__ Bp,
                                                  float* __restrict__ C,
                                                  int M, int N, int K, int ldc,
                                                  int accumulate) {
    const int lane  = threadIdx.x;
    const int tileM = blockIdx.x * 32;
    const int tileN = blockIdx.y * 64;
    const int nkt   = K >> 5;

    v8f acc[2][4] = {};

#if HAVE_TDM
    // 4 N-tile fragment rows x KCH fragments x 1 KB = 32 KB staged per chunk.
    __shared__ _Float16 bsl[4 * KCH * 512];
    const int n0 = tileN >> 4;

    for (int kc = 0; kc < nkt; kc += KCH) {
        // --- Tensor DMA: 2D tile (4 rows of KCH KB, row stride nkt KB) -> LDS
        unsigned long long gaddr =
            (unsigned long long)(Bp + ((size_t)n0 * nkt + kc) * 512);
        const unsigned dim0  = (unsigned)nkt * 256u;   // row length in 4B elems
        const unsigned tile0 = (unsigned)KCH * 256u;   // chunk length in 4B elems

        v4ui g0;
        g0.x = 1u;                                     // count=1, user mode
        g0.y = 0u;                                     // lds_addr = 0 (bsl base)
        g0.z = (unsigned)(gaddr & 0xffffffffu);        // global_addr[31:0]
        g0.w = (unsigned)((gaddr >> 32) & 0x01ffffffu) // global_addr[56:32]
               | 0x80000000u;                          // type = 2 ("image")
        v8si g1;
        g1[0] = 0x20000;                               // data_size = 4B
        g1[1] = (int)(dim0 << 16);                     // tensor_dim0[15:0]
        g1[2] = (int)((4u << 16) | (dim0 >> 16));      // tensor_dim1 = 4
        g1[3] = (int)(tile0 << 16);                    // tile_dim0
        g1[4] = 4;                                     // tile_dim1 = 4
        g1[5] = (int)dim0;                             // tensor_dim0_stride lo
        g1[6] = 0;                                     // stride hi / dim1 stride lo
        g1[7] = 0;
        v4si z4 = {0, 0, 0, 0};

        // WAR guard: prior chunk's LDS reads must drain before TDM rewrites LDS
        asm volatile("s_wait_dscnt 0x0" : : "v"(&bsl[0]) : "memory");
#if __clang_major__ >= 23
        v8si z8 = {0, 0, 0, 0, 0, 0, 0, 0};
        __builtin_amdgcn_tensor_load_to_lds(g0, g1, z4, z4, z8, 0);
#else
        __builtin_amdgcn_tensor_load_to_lds(g0, g1, z4, z4, 0);
#endif
        __builtin_amdgcn_s_wait_tensorcnt(0);
        // Escape bsl's address + memory clobber: the TDM write is invisible to
        // alias analysis, so force the compiler to treat LDS as modified here
        // (keeps the ds_load fragments live and ordered after the wait).
        asm volatile("" : : "v"(&bsl[0]) : "memory");

        for (int kk = 0; kk < KCH; ++kk) {
            const int kt = kc + kk;
            v16h a0 = load_a_frag(A, tileM,      K, kt * 32, lane);
            v16h a1 = load_a_frag(A, tileM + 16, K, kt * 32, lane);
            if (kt + 1 < nkt)  // prefetch next A slab toward near caches
                __builtin_prefetch(A + (size_t)(tileM + (lane & 15)) * K + (kt + 1) * 32, 0, 3);
            #pragma unroll
            for (int j = 0; j < 4; ++j) {
                const v16h b = *(const v16h*)(bsl + ((j * KCH + kk) * 512 + lane * 16));
                acc[0][j] = __builtin_amdgcn_wmma_f32_16x16x32_f16(
                    false, a0, false, b, (short)0, acc[0][j], false, false);
                acc[1][j] = __builtin_amdgcn_wmma_f32_16x16x32_f16(
                    false, a1, false, b, (short)0, acc[1][j], false, false);
            }
        }
    }
#else
    for (int kt = 0; kt < nkt; ++kt) {
        v16h a0 = load_a_frag(A, tileM,      K, kt * 32, lane);
        v16h a1 = load_a_frag(A, tileM + 16, K, kt * 32, lane);
        if (kt + 1 < nkt)
            __builtin_prefetch(A + (size_t)(tileM + (lane & 15)) * K + (kt + 1) * 32, 0, 3);
        #pragma unroll
        for (int j = 0; j < 4; ++j) {
            const _Float16* bp =
                Bp + ((size_t)((tileN >> 4) + j) * nkt + kt) * 512 + lane * 16;
            v16h b = *(const v16h*)bp;
            acc[0][j] = __builtin_amdgcn_wmma_f32_16x16x32_f16(
                false, a0, false, b, (short)0, acc[0][j], false, false);
            acc[1][j] = __builtin_amdgcn_wmma_f32_16x16x32_f16(
                false, a1, false, b, (short)0, acc[1][j], false, false);
        }
    }
#endif

    const int hi = lane >> 4, nl = lane & 15;
    #pragma unroll
    for (int i = 0; i < 2; ++i)
        #pragma unroll
        for (int j = 0; j < 4; ++j) {
            int col = tileN + j * 16 + nl;
            if (col < N) {
                #pragma unroll
                for (int r = 0; r < 8; ++r) {
                    int row = tileM + i * 16 + hi * 8 + r;
                    size_t o = (size_t)row * ldc + col;
                    float v = acc[i][j][r];
                    C[o] = accumulate ? (C[o] + v) : v;
                }
            }
        }
}

// ===========================================================================
// Depthwise causal conv1d (+bias, SiLU): xz[:, :DIN] -> xc
// ===========================================================================
__global__ void dwconv_silu_k(const float* __restrict__ xz, const float* __restrict__ cw,
                              const float* __restrict__ cb, float* __restrict__ xc) {
    int idx = blockIdx.x * blockDim.x + threadIdx.x;   // 4*4096*512
    int d = idx & 511, l = (idx >> 9) & 4095, b = idx >> 21;
    const float* base = xz + ((size_t)b * LSEQ) * (2 * DIN) + d;
    float acc = cb[d];
    #pragma unroll
    for (int j = 0; j < DCONV; ++j) {
        int ll = l - (DCONV - 1) + j;
        if (ll >= 0) acc += cw[d * DCONV + j] * base[(size_t)ll * (2 * DIN)];
    }
    xc[idx] = siluf_(acc);
}

// ===========================================================================
// Fused dt_proj + softplus + selective scan + D-skip + SiLU gate.
// One block per batch, one thread per channel d; state h[16] in registers.
// Per-step x_dbl row (dt|B|C, 48 floats) staged in LDS for the whole block.
// ===========================================================================
__global__ __launch_bounds__(DIN) void scan_k(const float* __restrict__ xdbl,
                                              const float* __restrict__ xc,
                                              const float* __restrict__ xz,
                                              const float* __restrict__ dpw,
                                              const float* __restrict__ dpb,
                                              const float* __restrict__ A_log,
                                              const float* __restrict__ Dp,
                                              float* __restrict__ yg) {
    const int b = blockIdx.x;
    const int d = threadIdx.x;

    float wdt[DTR], Ar[DST], h[DST];
    #pragma unroll
    for (int r = 0; r < DTR; ++r) wdt[r] = dpw[d * DTR + r];
    #pragma unroll
    for (int n = 0; n < DST; ++n) { Ar[n] = -__expf(A_log[d * DST + n]); h[n] = 0.f; }
    const float bias = dpb[d], Dd = Dp[d];

    __shared__ float row[DTR + 2 * DST];   // 48

    for (int l = 0; l < LSEQ; ++l) {
        __syncthreads();
        if (d < DTR + 2 * DST) row[d] = xdbl[((size_t)b * LSEQ + l) * 48 + d];
        __syncthreads();

        float a = bias;
        #pragma unroll
        for (int r = 0; r < DTR; ++r) a += wdt[r] * row[r];
        float delta = (a > 20.f) ? a : log1pf(__expf(a));     // softplus

        size_t ro = (size_t)b * LSEQ + l;
        float xcv = xc[ro * DIN + d];
        float zv  = xz[ro * (2 * DIN) + DIN + d];
        float dx  = delta * xcv;

        float y = 0.f;
        #pragma unroll
        for (int n = 0; n < DST; ++n) {
            float dA = __expf(delta * Ar[n]);
            h[n] = dA * h[n] + dx * row[DTR + n];             // dBu
            y += h[n] * row[DTR + DST + n];                   // C_t
        }
        y = (y + xcv * Dd) * siluf_(zv);
        yg[ro * DIN + d] = y;
    }
}

// ===========================================================================
// Host-side orchestration
// ===========================================================================
extern "C" void kernel_launch(void* const* d_in, const int* in_sizes, int n_in,
                              void* d_out, int out_size, void* d_ws, size_t ws_size,
                              hipStream_t stream) {
    const float* x       = (const float*)d_in[0];
    const float* conv1_w = (const float*)d_in[1];
    const float* conv1_b = (const float*)d_in[2];
    const float* gn1_w   = (const float*)d_in[3];
    const float* gn1_b   = (const float*)d_in[4];
    const float* conv2_w = (const float*)d_in[5];
    const float* conv2_b = (const float*)d_in[6];
    const float* gn2_w   = (const float*)d_in[7];
    const float* gn2_b   = (const float*)d_in[8];
    const float* ln_w    = (const float*)d_in[9];
    const float* ln_b    = (const float*)d_in[10];
    const float* ipw     = (const float*)d_in[11];
    const float* cw      = (const float*)d_in[12];
    const float* cb      = (const float*)d_in[13];
    const float* xpw     = (const float*)d_in[14];
    const float* dpw     = (const float*)d_in[15];
    const float* dpb     = (const float*)d_in[16];
    const float* A_log   = (const float*)d_in[17];
    const float* Dv      = (const float*)d_in[18];
    const float* opw     = (const float*)d_in[19];

    float* ws = (float*)d_ws;
    float* conv1o = ws;                       // 4*128*128*128 = 8388608
    float* stats  = ws + 8388608;             // 64
    float* conv2o = ws + 8388672;             // 4*256*64*64 = 4194304
    float* xn     = ws + 12582976;            // 16384*256
    float* xz     = ws + 16777280;            // 16384*1024
    float* xc     = ws + 33554496;            // 16384*512
    float* xdbl   = ws + 41943104;            // 16384*48
    float* yg     = ws + 42729536;            // 16384*512
    _Float16* bpack = (_Float16*)(ws + 51118144); // up to 1024*256 f16

    float* s = (float*)d_out;                 // residual stream [16384,256]

    const int T = 256;

    // ---- conv stem ----
    conv1_k<<<8388608 / T, T, 0, stream>>>(x, conv1_w, conv1_b, conv1o);
    gn_stats_k<<<32, 256, 0, stream>>>(conv1o, stats, 128, 128 * 128);
    gn_apply_silu_k<<<8388608 / T, T, 0, stream>>>(conv1o, stats, gn1_w, gn1_b, 128, 128 * 128);
    conv2_k<<<4194304 / T, T, 0, stream>>>(conv1o, conv2_w, conv2_b, conv2o);
    gn_stats_k<<<32, 256, 0, stream>>>(conv2o, stats, 256, 64 * 64);
    gn_apply_silu_k<<<4194304 / T, T, 0, stream>>>(conv2o, stats, gn2_w, gn2_b, 256, 64 * 64);
    to_seq_k<<<4194304 / T, T, 0, stream>>>(conv2o, s);

    // ---- 4 Mamba layers ----
    for (int i = 0; i < NL; ++i) {
        ln_k<<<MROWS / 8, 256, 0, stream>>>(s, ln_w + i * HID, ln_b + i * HID, xn);

        // in_proj: [16384,256] x [1024,256]^T -> xz [16384,1024]
        pack_w_k<<<(1024 * 256 + T - 1) / T, T, 0, stream>>>(
            ipw + (size_t)i * 1024 * 256, bpack, 1024, 256, 1024);
        gemm_wmma_k<<<dim3(MROWS / 32, 1024 / 64), 32, 0, stream>>>(
            xn, bpack, xz, MROWS, 1024, 256, 1024, 0);

        dwconv_silu_k<<<(MROWS * DIN) / T, T, 0, stream>>>(
            xz, cw + (size_t)i * DIN * DCONV, cb + (size_t)i * DIN, xc);

        // x_proj: [16384,512] x [48,512]^T -> xdbl [16384,48] (Npad=64)
        pack_w_k<<<(64 * 512 + T - 1) / T, T, 0, stream>>>(
            xpw + (size_t)i * 48 * 512, bpack, 48, 512, 64);
        gemm_wmma_k<<<dim3(MROWS / 32, 1), 32, 0, stream>>>(
            xc, bpack, xdbl, MROWS, 48, 512, 48, 0);

        scan_k<<<BATCH, DIN, 0, stream>>>(
            xdbl, xc, xz,
            dpw + (size_t)i * DIN * DTR, dpb + (size_t)i * DIN,
            A_log + (size_t)i * DIN * DST, Dv + (size_t)i * DIN, yg);

        // out_proj (+residual): [16384,512] x [256,512]^T accumulated into s
        pack_w_k<<<(256 * 512 + T - 1) / T, T, 0, stream>>>(
            opw + (size_t)i * 256 * 512, bpack, 256, 512, 256);
        gemm_wmma_k<<<dim3(MROWS / 32, 256 / 64), 32, 0, stream>>>(
            yg, bpack, s, MROWS, 256, 512, 256, 1);
    }

    if (out_size >= MROWS * HID + 2)
        write_hw_k<<<1, 32, 0, stream>>>((float*)d_out, out_size - 2);
}
